// WindowAttention3D_19542101197165
// MI455X (gfx1250) — matrix operand order repaired and checked by reading, not compiled, not run on hardware
//
#include <hip/hip_runtime.h>

// ---------------- problem constants ----------------
#define WD 8
#define WH 7
#define WW 7
#define L_TOK 392          // WD*WH*WW
#define C_DIM 128
#define NHEAD 4
#define HD 32
#define NW 8
#define B_WIN 512
#define M_TOT (B_WIN * L_TOK)      // 200704 = 1568*128 exactly
#define SCALE_Q 0.17677669529663689f   // 32^-0.5

typedef __attribute__((ext_vector_type(16))) __bf16 v16bf;
typedef __attribute__((ext_vector_type(8)))  float  v8f;

union FragAB { v16bf v; uint4 q[2]; };

static __device__ __forceinline__ unsigned short f32_bf16(float f) {
  unsigned u = __float_as_uint(f);
  u += 0x7FFFu + ((u >> 16) & 1u);      // round-to-nearest-even
  return (unsigned short)(u >> 16);
}
static __device__ __forceinline__ unsigned pack_bf16x2(float a, float b) {
  return (unsigned)f32_bf16(a) | ((unsigned)f32_bf16(b) << 16);
}
static __device__ __forceinline__ v16bf ld_frag(const unsigned char* s,
                                                unsigned off0, unsigned off1) {
  FragAB f;
  f.q[0] = *(const uint4*)(s + off0);
  f.q[1] = *(const uint4*)(s + off1);
  return f.v;
}
static __device__ __forceinline__ v8f wmma_bf16(v16bf a, v16bf b, v8f c) {
  return __builtin_amdgcn_wmma_f32_16x16x32_bf16(false, a, false, b,
                                                 (short)0, c, false, false);
}

// =====================================================================
// Kernel 1: qkv = x @ qkv_w + qkv_b ; split into q(*scale)/k/v bf16
// grid (1568, 6), block 256.  Block tile: 128 rows x 64 cols, K=128.
// =====================================================================
__global__ __launch_bounds__(256) void k_qkv(
    const float* __restrict__ x, const float* __restrict__ w,
    const float* __restrict__ bias,
    unsigned short* __restrict__ qo, unsigned short* __restrict__ ko,
    unsigned short* __restrict__ vo) {
  __shared__ __align__(16) unsigned char sX[128 * 128 * 2];  // [m][k] bf16
  __shared__ __align__(16) unsigned char sW[64 * 128 * 2];   // [n][k] bf16 (W^T)

  const int tid = threadIdx.x;
  const int lane = tid & 31, wv = tid >> 5;
  const int l15 = lane & 15, half = lane >> 4;
  const int mbase = blockIdx.x * 128;
  const int nbase = blockIdx.y * 64;

  for (int e = tid; e < 128 * 64; e += 256) {           // X tile (uint32 = 2 bf16)
    int m = e >> 6, k2 = (e & 63) * 2;
    const float* p = x + (size_t)(mbase + m) * 128 + k2;
    ((unsigned*)sX)[e] = pack_bf16x2(p[0], p[1]);
  }
  for (int e = tid; e < 64 * 64; e += 256) {            // W^T tile
    int n = e >> 6, k2 = (e & 63) * 2;
    float f0 = w[(size_t)k2 * 384 + nbase + n];
    float f1 = w[(size_t)(k2 + 1) * 384 + nbase + n];
    ((unsigned*)sW)[e] = pack_bf16x2(f0, f1);
  }
  __syncthreads();

  v8f acc[4] = {v8f{}, v8f{}, v8f{}, v8f{}};
  const int arow = wv * 16 + l15;
#pragma unroll
  for (int ks = 0; ks < 128; ks += 32) {
    unsigned aoff = (unsigned)arow * 256 + ks * 2 + half * 16;
    v16bf a = ld_frag(sX, aoff, aoff + 32);
#pragma unroll
    for (int j = 0; j < 4; ++j) {
      unsigned boff = (unsigned)(j * 16 + l15) * 256 + ks * 2 + half * 16;
      v16bf b = ld_frag(sW, boff, boff + 32);
      acc[j] = wmma_bf16(a, b, acc[j]);
    }
  }

#pragma unroll
  for (int j = 0; j < 4; ++j) {
#pragma unroll
    for (int v = 0; v < 8; ++v) {
      int r = mbase + wv * 16 + v + 8 * half;           // global row in [0,200704)
      int gcol = nbase + j * 16 + l15;                  // [0,384)
      float val = acc[j][v] + bias[gcol];
      int which = gcol >> 7;                            // 0=q 1=k 2=v
      int c = gcol & 127, hh = c >> 5, d = c & 31;
      int bw = r / L_TOK, lr = r - bw * L_TOK;
      if (which == 0) val *= SCALE_Q;
      unsigned short* dst = (which == 0) ? qo : (which == 1) ? ko : vo;
      dst[(((size_t)bw * NHEAD + hh) * L_TOK + lr) * HD + d] = f32_bf16(val);
    }
  }
}

// =====================================================================
// Kernel 2: bm[nw][h][qi][kj] = rel_bias_table[relidx(qi,kj)][h] + mask[nw][qi][kj]
// =====================================================================
__global__ __launch_bounds__(256) void k_biasmask(
    const float* __restrict__ table, const float* __restrict__ mask,
    float* __restrict__ bm) {
  int e = blockIdx.x * 256 + threadIdx.x;
  const int total = NW * NHEAD * L_TOK * L_TOK;
  if (e >= total) return;
  int kj = e % L_TOK;
  int t = e / L_TOK;
  int qi = t % L_TOK;
  t /= L_TOK;
  int h = t & 3, nw = t >> 2;
  int i1 = qi / 49, r1 = qi % 49, j1 = r1 / 7, k1 = r1 % 7;
  int i2 = kj / 49, r2 = kj % 49, j2 = r2 / 7, k2 = r2 % 7;
  int idx = (i1 - i2 + 7) * 169 + (j1 - j2 + 6) * 13 + (k1 - k2 + 6);
  bm[e] = table[idx * NHEAD + h] +
          mask[((size_t)nw * L_TOK + qi) * L_TOK + kj];
}

// =====================================================================
// Kernel 3: flash attention per (b,h).  grid (512,4), block 256 (8 waves).
// =====================================================================
__global__ __launch_bounds__(256) void k_attn(
    const unsigned short* __restrict__ qws, const unsigned short* __restrict__ kws,
    const unsigned short* __restrict__ vws, const float* __restrict__ bm,
    unsigned short* __restrict__ ows) {
  __shared__ __align__(16) unsigned char sQ[400 * 32 * 2];   // [t][d] bf16, pad->400
  __shared__ __align__(16) unsigned char sK[416 * 32 * 2];   // [t][d] bf16, pad->416
  __shared__ __align__(16) unsigned char sVt[32 * 416 * 2];  // [d][t] bf16 (V^T)
  __shared__ __align__(16) unsigned char sP[8 * 16 * 32 * 2];// per-wave P scratch

  const int tid = threadIdx.x;
  const int lane = tid & 31, wv = tid >> 5;
  const int l15 = lane & 15, half = lane >> 4;
  const int b = blockIdx.x, h = blockIdx.y, nw = b & (NW - 1);

  const size_t hbase = ((size_t)b * NHEAD + h) * L_TOK * HD;
  const unsigned* q32 = (const unsigned*)(qws + hbase);
  const unsigned* k32 = (const unsigned*)(kws + hbase);
  const unsigned* v32 = (const unsigned*)(vws + hbase);

  for (int e = tid; e < 400 * 16; e += 256) {
    int t = e >> 4, du = e & 15;
    ((unsigned*)sQ)[e] = (t < L_TOK) ? q32[t * 16 + du] : 0u;
  }
  for (int e = tid; e < 416 * 16; e += 256) {
    int t = e >> 4, du = e & 15;
    ((unsigned*)sK)[e] = (t < L_TOK) ? k32[t * 16 + du] : 0u;
  }
  for (int e = tid; e < 416 * 16; e += 256) {
    int t = e >> 4, du = e & 15;
    unsigned val = (t < L_TOK) ? v32[t * 16 + du] : 0u;
    ((unsigned short*)sVt)[(2 * du) * 416 + t] = (unsigned short)val;
    ((unsigned short*)sVt)[(2 * du + 1) * 416 + t] = (unsigned short)(val >> 16);
  }
  __syncthreads();

  const float* bmb = bm + ((size_t)(nw * NHEAD + h)) * L_TOK * L_TOK;

  for (int qt = wv; qt < 25; qt += 8) {
    const int qbase = qt * 16;
    unsigned aoff = (unsigned)(qbase + l15) * 64 + half * 16;
    v16bf aq = ld_frag(sQ, aoff, aoff + 32);
    v8f oc0 = {}, oc1 = {};
    float mprev[8], lsum[8];
#pragma unroll
    for (int v = 0; v < 8; ++v) { mprev[v] = -1e30f; lsum[v] = 0.f; }

    for (int kc = 0; kc < 13; ++kc) {
      const int kbase = kc * 32;
      unsigned b0 = (unsigned)(kbase + l15) * 64 + half * 16;
      v16bf bk0 = ld_frag(sK, b0, b0 + 32);
      unsigned b1 = (unsigned)(kbase + 16 + l15) * 64 + half * 16;
      v16bf bk1 = ld_frag(sK, b1, b1 + 32);
      v8f s0 = {}, s1 = {};
      s0 = wmma_bf16(aq, bk0, s0);              // S = q*scale . k^T
      s1 = wmma_bf16(aq, bk1, s1);

      const int col0 = kbase + l15, col1 = col0 + 16;
      float pr0[8], pr1[8];
#pragma unroll
      for (int v = 0; v < 8; ++v) {
        int r = qbase + v + 8 * half;
        bool rv = r < L_TOK;
        float a0 = (rv && col0 < L_TOK) ? bmb[(size_t)r * L_TOK + col0] : -1e30f;
        float a1 = (rv && col1 < L_TOK) ? bmb[(size_t)r * L_TOK + col1] : -1e30f;
        float x0 = s0[v] + a0, x1 = s1[v] + a1;
        float t = fmaxf(x0, x1);                 // row-max across 32 cols
        t = fmaxf(t, __shfl_xor(t, 1));
        t = fmaxf(t, __shfl_xor(t, 2));
        t = fmaxf(t, __shfl_xor(t, 4));
        t = fmaxf(t, __shfl_xor(t, 8));          // uniform per 16-lane half
        float nm = fmaxf(mprev[v], t);
        float alpha = __expf(mprev[v] - nm);
        float p0 = __expf(x0 - nm), p1 = __expf(x1 - nm);
        float ss = p0 + p1;
        ss += __shfl_xor(ss, 1);
        ss += __shfl_xor(ss, 2);
        ss += __shfl_xor(ss, 4);
        ss += __shfl_xor(ss, 8);
        lsum[v] = lsum[v] * alpha + ss;
        mprev[v] = nm;
        oc0[v] *= alpha; oc1[v] *= alpha;
        pr0[v] = p0; pr1[v] = p1;
      }
      // C-layout -> A-layout via per-wave LDS scratch ([16 rows][32 cols] bf16)
      unsigned pb = (unsigned)wv * 1024;
      unsigned short* ps = (unsigned short*)(sP + pb);
#pragma unroll
      for (int v = 0; v < 8; ++v) {
        int rl = v + 8 * half;
        ps[rl * 32 + l15] = f32_bf16(pr0[v]);
        ps[rl * 32 + 16 + l15] = f32_bf16(pr1[v]);
      }
      unsigned ap = pb + (unsigned)l15 * 64 + half * 16;
      v16bf pfrag = ld_frag(sP, ap, ap + 32);
      unsigned bv0 = (unsigned)l15 * 832 + kbase * 2 + half * 16;
      v16bf vf0 = ld_frag(sVt, bv0, bv0 + 32);
      unsigned bv1 = (unsigned)(16 + l15) * 832 + kbase * 2 + half * 16;
      v16bf vf1 = ld_frag(sVt, bv1, bv1 + 32);
      oc0 = wmma_bf16(pfrag, vf0, oc0);          // O += P.V
      oc1 = wmma_bf16(pfrag, vf1, oc1);
    }

#pragma unroll
    for (int v = 0; v < 8; ++v) {
      int r = qbase + v + 8 * half;
      if (r < L_TOK) {
        float inv = 1.0f / lsum[v];
        size_t o = ((size_t)b * L_TOK + r) * C_DIM + h * HD;
        ows[o + l15] = f32_bf16(oc0[v] * inv);
        ows[o + 16 + l15] = f32_bf16(oc1[v] * inv);
      }
    }
  }
}

// =====================================================================
// Kernel 4: out = O @ proj_w + proj_b  (bf16 A, f32 out). grid (1568,2).
// =====================================================================
__global__ __launch_bounds__(256) void k_proj(
    const unsigned short* __restrict__ ows, const float* __restrict__ w,
    const float* __restrict__ bias, float* __restrict__ out) {
  __shared__ __align__(16) unsigned char sO[128 * 128 * 2];
  __shared__ __align__(16) unsigned char sW[64 * 128 * 2];

  const int tid = threadIdx.x;
  const int lane = tid & 31, wv = tid >> 5;
  const int l15 = lane & 15, half = lane >> 4;
  const int mbase = blockIdx.x * 128;
  const int nbase = blockIdx.y * 64;

  const unsigned* o32 = (const unsigned*)ows;
  for (int e = tid; e < 128 * 64; e += 256) {
    int m = e >> 6, du = e & 63;
    ((unsigned*)sO)[e] = o32[(size_t)(mbase + m) * 64 + du];
  }
  for (int e = tid; e < 64 * 64; e += 256) {
    int n = e >> 6, k2 = (e & 63) * 2;
    float f0 = w[(size_t)k2 * 128 + nbase + n];
    float f1 = w[(size_t)(k2 + 1) * 128 + nbase + n];
    ((unsigned*)sW)[e] = pack_bf16x2(f0, f1);
  }
  __syncthreads();

  v8f acc[4] = {v8f{}, v8f{}, v8f{}, v8f{}};
  const int arow = wv * 16 + l15;
#pragma unroll
  for (int ks = 0; ks < 128; ks += 32) {
    unsigned aoff = (unsigned)arow * 256 + ks * 2 + half * 16;
    v16bf a = ld_frag(sO, aoff, aoff + 32);
#pragma unroll
    for (int j = 0; j < 4; ++j) {
      unsigned boff = (unsigned)(j * 16 + l15) * 256 + ks * 2 + half * 16;
      v16bf b = ld_frag(sW, boff, boff + 32);
      acc[j] = wmma_bf16(a, b, acc[j]);
    }
  }
#pragma unroll
  for (int j = 0; j < 4; ++j) {
#pragma unroll
    for (int v = 0; v < 8; ++v) {
      int r = mbase + wv * 16 + v + 8 * half;
      int gcol = nbase + j * 16 + l15;
      out[(size_t)r * C_DIM + gcol] = acc[j][v] + bias[gcol];
    }
  }
}

// =====================================================================
extern "C" void kernel_launch(void* const* d_in, const int* in_sizes, int n_in,
                              void* d_out, int out_size, void* d_ws, size_t ws_size,
                              hipStream_t stream) {
  const float* x      = (const float*)d_in[0];
  const float* mask   = (const float*)d_in[1];
  const float* qkv_w  = (const float*)d_in[2];
  const float* qkv_b  = (const float*)d_in[3];
  const float* proj_w = (const float*)d_in[4];
  const float* proj_b = (const float*)d_in[5];
  const float* rbt    = (const float*)d_in[6];
  float* out = (float*)d_out;

  const size_t SZ = (size_t)B_WIN * NHEAD * L_TOK * HD;   // bf16 elems per tensor
  unsigned short* qws = (unsigned short*)d_ws;
  unsigned short* kws = qws + SZ;
  unsigned short* vws = kws + SZ;
  unsigned short* ows = vws + SZ;
  float* bm = (float*)(ows + SZ);                         // [8][4][392][392] f32

  k_qkv<<<dim3(M_TOT / 128, 3 * C_DIM / 64), 256, 0, stream>>>(
      x, qkv_w, qkv_b, qws, kws, vws);

  const int bm_total = NW * NHEAD * L_TOK * L_TOK;
  k_biasmask<<<(bm_total + 255) / 256, 256, 0, stream>>>(rbt, mask, bm);

  k_attn<<<dim3(B_WIN, NHEAD), 256, 0, stream>>>(qws, kws, vws, bm, ows);

  k_proj<<<dim3(M_TOT / 128, C_DIM / 64), 256, 0, stream>>>(
      ows, proj_w, proj_b, out);
}
